// Transformer_XA_78666620993955
// MI455X (gfx1250) — matrix operand order repaired and checked by reading, not compiled, not run on hardware
//
#include <hip/hip_runtime.h>
#include <math.h>

// ---------------------------------------------------------------------------
// Problem dims (fixed by the reference)
#define N_ 256
#define M_ 512
#define S_ 128
#define D_ 512

typedef __attribute__((ext_vector_type(16))) __bf16 v16bf;
typedef __attribute__((ext_vector_type(8)))  float  v8f;

// ---------------------------------------------------------------------------
// WMMA helper (CDNA5 wave32, D = A(16x32) * B(32x16) + C, f32 accum)
__device__ __forceinline__ v8f wmma_bf16(v16bf a, v16bf b, v8f c) {
  return __builtin_amdgcn_wmma_f32_16x16x32_bf16(false, a, false, b,
                                                 (short)0, c, false, false);
}

// CDNA5 async memory->LDS copy (16B per lane), tracked with ASYNCcnt.
// lds_off is the 32-bit LDS byte offset (low 32 bits of the flat shared addr
// per the gfx1250 LDS aperture mapping), gaddr a per-lane global address.
__device__ __forceinline__ void async_ld_b128(uint32_t lds_off, uint64_t gaddr) {
  asm volatile("global_load_async_to_lds_b128 %0, %1, off"
               :: "v"(lds_off), "v"(gaddr) : "memory");
}
__device__ __forceinline__ void wait_async0() {
  asm volatile("s_wait_asynccnt 0x0" ::: "memory");
}

// Stage `bytes` (multiple of 4096) from gsrc into LDS at lbase; 256 threads.
__device__ __forceinline__ void stage_tile_async(uint32_t lbase, const void* gsrc,
                                                 int bytes, int tid) {
  const uint64_t g = (uint64_t)gsrc;
  for (int i = tid; i < (bytes >> 4); i += 256)
    async_ld_b128(lbase + (uint32_t)(i << 4), g + ((uint64_t)i << 4));
  wait_async0();
}

// A fragment: 16x32 bf16 tile, row-major source, leading dim ld.
__device__ __forceinline__ v16bf frag_a(const __bf16* src, int ld) {
  const int lane = threadIdx.x & 31;
  const int row = lane & 15, hf = lane >> 4;
  v16bf f;
#pragma unroll
  for (int e = 0; e < 16; ++e) f[e] = src[row * ld + hf * 16 + e];
  return f;
}

// B fragment where B[k][n] = X[n][k] (X row-major, ld). Contiguous 32B/lane.
// Used for x @ W^T, q @ k^T, and weights @ V via the transposed-V layout.
__device__ __forceinline__ v16bf frag_bT(const __bf16* src, int ld) {
  const int lane = threadIdx.x & 31;
  const int col = lane & 15, hf = lane >> 4;
  v16bf f;
#pragma unroll
  for (int e = 0; e < 16; ++e) f[e] = src[col * ld + hf * 16 + e];
  return f;
}

// ---------------------------------------------------------------------------
// LayerNorm(f32 in) -> bf16 out, one row of D_=512 per 256-thread block.
__global__ __launch_bounds__(256) void ln_to_bf16(const float* __restrict__ X,
                                                  const float* __restrict__ g,
                                                  const float* __restrict__ b,
                                                  __bf16* __restrict__ Y) {
  __shared__ float sS[8], sQ[8];
  const size_t row = blockIdx.x;
  const int tid = threadIdx.x;
  const float x0 = X[row * D_ + tid];
  const float x1 = X[row * D_ + 256 + tid];
  float s = x0 + x1, q = x0 * x0 + x1 * x1;
#pragma unroll
  for (int o = 1; o < 32; o <<= 1) {
    s += __shfl_xor(s, o, 32);
    q += __shfl_xor(q, o, 32);
  }
  const int w = tid >> 5, lane = tid & 31;
  if (lane == 0) { sS[w] = s; sQ[w] = q; }
  __syncthreads();
  s = 0.f; q = 0.f;
#pragma unroll
  for (int i = 0; i < 8; ++i) { s += sS[i]; q += sQ[i]; }
  const float mean = s * (1.f / D_);
  const float var  = q * (1.f / D_) - mean * mean;
  const float rstd = rsqrtf(var + 1e-5f);
  Y[row * D_ + tid]       = (__bf16)((x0 - mean) * rstd * g[tid] + b[tid]);
  Y[row * D_ + 256 + tid] = (__bf16)((x1 - mean) * rstd * g[256 + tid] + b[256 + tid]);
}

// f32 -> bf16 elementwise (weight conversion)
__global__ void cvt_bf16(const float* __restrict__ x, __bf16* __restrict__ y, int n) {
  int i = blockIdx.x * blockDim.x + threadIdx.x;
  if (i < n) y[i] = (__bf16)x[i];
}

// ---------------------------------------------------------------------------
// C = A(bf16, R x 512) @ W^T(bf16, 512 x 512) + bias(f32), output bf16.
// One block = 32 rows x 512 cols; 8 waves, wave w owns cols [w*64, w*64+64).
// transpose_out == 0: C row-major (R x 512).
// transpose_out != 0: rows are (m, s) with s in [0,S_); write per-m transposed
//                     layout C[m][d][s] (leading dim S_) for the attention V.
__global__ __launch_bounds__(256) void gemm_xwT(const __bf16* __restrict__ A,
                                                const __bf16* __restrict__ W,
                                                const float* __restrict__ bias,
                                                __bf16* __restrict__ C,
                                                int transpose_out) {
  __shared__ __bf16 atile[32][D_];  // 32KB
  const int tid = threadIdx.x, w = tid >> 5, lane = tid & 31;
  const int hf = lane >> 4, lc = lane & 15;
  const size_t row0 = (size_t)blockIdx.x * 32;

  // async A-tile stage: global -> LDS directly (no VGPR round trip)
  stage_tile_async((uint32_t)(size_t)(void*)&atile[0][0], A + row0 * D_,
                   32 * D_ * 2, tid);
  __syncthreads();

  const int d0 = w * 64;
  v8f acc[2][4] = {};
  for (int kk = 0; kk < D_; kk += 32) {
    v16bf a0 = frag_a(&atile[0][kk], D_);
    v16bf a1 = frag_a(&atile[16][kk], D_);
#pragma unroll
    for (int ct = 0; ct < 4; ++ct) {
      v16bf b = frag_bT(W + (size_t)(d0 + ct * 16) * D_ + kk, D_);
      acc[0][ct] = wmma_bf16(a0, b, acc[0][ct]);
      acc[1][ct] = wmma_bf16(a1, b, acc[1][ct]);
    }
  }
  if (transpose_out == 0) {
#pragma unroll
    for (int rt = 0; rt < 2; ++rt)
#pragma unroll
      for (int ct = 0; ct < 4; ++ct) {
        const int col = d0 + ct * 16 + lc;
        const float bb = bias[col];
#pragma unroll
        for (int r = 0; r < 8; ++r) {
          const int row = rt * 16 + hf * 8 + r;
          C[(row0 + row) * D_ + col] = (__bf16)(acc[rt][ct][r] + bb);
        }
      }
  } else {
    const size_t m = row0 >> 7;          // row0 / S_
    const int sloc = (int)(row0 & (S_ - 1));
    __bf16* Cm = C + m * (size_t)D_ * S_;
#pragma unroll
    for (int rt = 0; rt < 2; ++rt)
#pragma unroll
      for (int ct = 0; ct < 4; ++ct) {
        const int col = d0 + ct * 16 + lc;
        const float bb = bias[col];
#pragma unroll
        for (int r = 0; r < 8; ++r) {
          const int row = rt * 16 + hf * 8 + r;
          Cm[(size_t)col * S_ + sloc + row] = (__bf16)(acc[rt][ct][r] + bb);
        }
      }
  }
}

// ---------------------------------------------------------------------------
// Fused attention + output head.
// grid = (M_, N_/32); block = 256 (8 wave32).
//  1) logits(32x128) = q @ k_m^T / sqrt(D)      (wave w: S-cols w*16..)
//  2) masked softmax over S
//  3) attn(32x512)  = weights @ v_m (VT layout) (wave w: D-cols w*64..)
//  4) o = attn @ Wo^T + bo ; LN2 (cross-wave stats)
//  5) lin = o_ln @ Wl^T + bl ; out = LN3(o_ln + lin)  -> global
__global__ __launch_bounds__(256) void attn_fused(
    const __bf16* __restrict__ qbf, const __bf16* __restrict__ kbf,
    const __bf16* __restrict__ vtbf, const int* __restrict__ mask,
    const __bf16* __restrict__ Wo, const float* __restrict__ bo,
    const __bf16* __restrict__ Wl, const float* __restrict__ bl,
    const float* __restrict__ g2, const float* __restrict__ b2,
    const float* __restrict__ g3, const float* __restrict__ b3,
    float* __restrict__ out) {
  __shared__ float  logits[32][S_];   // 16KB
  __shared__ __bf16 wts[32][S_];      //  8KB
  __shared__ __bf16 abuf[32][D_];     // 32KB: q-tile -> attn-tile -> oln-tile
  __shared__ float  redS[8][32], redQ[8][32];  // 2KB LN partials

  const int m = blockIdx.x;
  const int n0 = blockIdx.y * 32;
  const int tid = threadIdx.x, w = tid >> 5, lane = tid & 31;
  const int hf = lane >> 4, lc = lane & 15;
  const int d0 = w * 64;

  // async stage of q rows [n0, n0+32) into abuf
  stage_tile_async((uint32_t)(size_t)(void*)&abuf[0][0],
                   qbf + (size_t)n0 * D_, 32 * D_ * 2, tid);
  __syncthreads();

  // ---- Stage 1: logits ----
  {
    const int s0 = w * 16;
    v8f acc0 = {}, acc1 = {};
    const __bf16* kb = kbf + ((size_t)m * S_ + s0) * D_;
    for (int kk = 0; kk < D_; kk += 32) {
      v16bf a0 = frag_a(&abuf[0][kk], D_);
      v16bf a1 = frag_a(&abuf[16][kk], D_);
      v16bf bm = frag_bT(kb + kk, D_);
      acc0 = wmma_bf16(a0, bm, acc0);
      acc1 = wmma_bf16(a1, bm, acc1);
    }
#pragma unroll
    for (int r = 0; r < 8; ++r) {
      logits[hf * 8 + r][s0 + lc]      = acc0[r];
      logits[16 + hf * 8 + r][s0 + lc] = acc1[r];
    }
  }
  __syncthreads();

  // ---- Stage 2: masked softmax over S; wave w owns rows w*4..w*4+3 ----
  {
    const float scale = 0.04419417382415922f;  // 1/sqrt(512)
#pragma unroll
    for (int rr = 0; rr < 4; ++rr) {
      const int row = w * 4 + rr;
      float v0[4];
#pragma unroll
      for (int j = 0; j < 4; ++j) {
        const int c = lane * 4 + j;
        float l = logits[row][c] * scale;
        if (mask[m * S_ + c] == 0) l = -__builtin_inff();
        v0[j] = l;
      }
      float mx = fmaxf(fmaxf(v0[0], v0[1]), fmaxf(v0[2], v0[3]));
#pragma unroll
      for (int o = 1; o < 32; o <<= 1) mx = fmaxf(mx, __shfl_xor(mx, o, 32));
      float ex[4], sum = 0.f;
#pragma unroll
      for (int j = 0; j < 4; ++j) { ex[j] = __expf(v0[j] - mx); sum += ex[j]; }
#pragma unroll
      for (int o = 1; o < 32; o <<= 1) sum += __shfl_xor(sum, o, 32);
      const float inv = 1.f / sum;
#pragma unroll
      for (int j = 0; j < 4; ++j) wts[row][lane * 4 + j] = (__bf16)(ex[j] * inv);
    }
  }
  __syncthreads();

  // ---- Stage 3: attn = weights @ v_m (V stored [m][d][s], ld = S_) ----
  v8f acc[2][4] = {};
  {
    const __bf16* vt = vtbf + (size_t)m * D_ * S_;
    for (int kk = 0; kk < S_; kk += 32) {
      v16bf a0 = frag_a(&wts[0][kk], S_);
      v16bf a1 = frag_a(&wts[16][kk], S_);
#pragma unroll
      for (int ct = 0; ct < 4; ++ct) {
        v16bf bm = frag_bT(vt + (size_t)(d0 + ct * 16) * S_ + kk, S_);
        acc[0][ct] = wmma_bf16(a0, bm, acc[0][ct]);
        acc[1][ct] = wmma_bf16(a1, bm, acc[1][ct]);
      }
    }
  }
  // spill attn tile to LDS (bf16) as A-operand for o-projection
#pragma unroll
  for (int rt = 0; rt < 2; ++rt)
#pragma unroll
    for (int ct = 0; ct < 4; ++ct)
#pragma unroll
      for (int r = 0; r < 8; ++r)
        abuf[rt * 16 + hf * 8 + r][d0 + ct * 16 + lc] = (__bf16)acc[rt][ct][r];
  __syncthreads();

  // cross-wave LayerNorm over the 32x512 tile held in acc registers
  auto layernorm_tile = [&](v8f(&A)[2][4], const float* g, const float* b) {
#pragma unroll
    for (int rt = 0; rt < 2; ++rt)
#pragma unroll
      for (int r = 0; r < 8; ++r) {
        float s = 0.f, q = 0.f;
#pragma unroll
        for (int ct = 0; ct < 4; ++ct) { float v = A[rt][ct][r]; s += v; q += v * v; }
#pragma unroll
        for (int o = 1; o < 16; o <<= 1) {  // reduce within 16-lane half
          s += __shfl_xor(s, o, 32);
          q += __shfl_xor(q, o, 32);
        }
        if (lc == 0) {
          const int row = rt * 16 + hf * 8 + r;
          redS[w][row] = s; redQ[w][row] = q;
        }
      }
    __syncthreads();
#pragma unroll
    for (int rt = 0; rt < 2; ++rt)
#pragma unroll
      for (int r = 0; r < 8; ++r) {
        const int row = rt * 16 + hf * 8 + r;
        float s = 0.f, q = 0.f;
#pragma unroll
        for (int ww = 0; ww < 8; ++ww) { s += redS[ww][row]; q += redQ[ww][row]; }
        const float mean = s * (1.f / D_);
        const float var  = q * (1.f / D_) - mean * mean;
        const float rstd = rsqrtf(var + 1e-5f);
#pragma unroll
        for (int ct = 0; ct < 4; ++ct) {
          const int col = d0 + ct * 16 + lc;
          A[rt][ct][r] = (A[rt][ct][r] - mean) * rstd * g[col] + b[col];
        }
      }
    __syncthreads();
  };

  // ---- Stage 4: o = attn @ Wo^T + bo ; LN2 ----
#pragma unroll
  for (int rt = 0; rt < 2; ++rt)
#pragma unroll
    for (int ct = 0; ct < 4; ++ct) acc[rt][ct] = (v8f){};
  for (int kk = 0; kk < D_; kk += 32) {
    v16bf a0 = frag_a(&abuf[0][kk], D_);
    v16bf a1 = frag_a(&abuf[16][kk], D_);
#pragma unroll
    for (int ct = 0; ct < 4; ++ct) {
      v16bf bm = frag_bT(Wo + (size_t)(d0 + ct * 16) * D_ + kk, D_);
      acc[0][ct] = wmma_bf16(a0, bm, acc[0][ct]);
      acc[1][ct] = wmma_bf16(a1, bm, acc[1][ct]);
    }
  }
#pragma unroll
  for (int rt = 0; rt < 2; ++rt)
#pragma unroll
    for (int ct = 0; ct < 4; ++ct) {
      const float bb = bo[d0 + ct * 16 + lc];
#pragma unroll
      for (int r = 0; r < 8; ++r) acc[rt][ct][r] += bb;
    }
  layernorm_tile(acc, g2, b2);  // acc now holds o_ln (f32, kept live)

  // oln (bf16) -> abuf as A-operand for the final linear
#pragma unroll
  for (int rt = 0; rt < 2; ++rt)
#pragma unroll
    for (int ct = 0; ct < 4; ++ct)
#pragma unroll
      for (int r = 0; r < 8; ++r)
        abuf[rt * 16 + hf * 8 + r][d0 + ct * 16 + lc] = (__bf16)acc[rt][ct][r];
  __syncthreads();

  // ---- Stage 5: lin = o_ln @ Wl^T + bl ; out = LN3(o_ln + lin) ----
  v8f lacc[2][4] = {};
  for (int kk = 0; kk < D_; kk += 32) {
    v16bf a0 = frag_a(&abuf[0][kk], D_);
    v16bf a1 = frag_a(&abuf[16][kk], D_);
#pragma unroll
    for (int ct = 0; ct < 4; ++ct) {
      v16bf bm = frag_bT(Wl + (size_t)(d0 + ct * 16) * D_ + kk, D_);
      lacc[0][ct] = wmma_bf16(a0, bm, lacc[0][ct]);
      lacc[1][ct] = wmma_bf16(a1, bm, lacc[1][ct]);
    }
  }
#pragma unroll
  for (int rt = 0; rt < 2; ++rt)
#pragma unroll
    for (int ct = 0; ct < 4; ++ct) {
      const float bb = bl[d0 + ct * 16 + lc];
#pragma unroll
      for (int r = 0; r < 8; ++r) acc[rt][ct][r] += lacc[rt][ct][r] + bb;  // residual
    }
  layernorm_tile(acc, g3, b3);

#pragma unroll
  for (int rt = 0; rt < 2; ++rt)
#pragma unroll
    for (int ct = 0; ct < 4; ++ct) {
      const int col = d0 + ct * 16 + lc;
#pragma unroll
      for (int r = 0; r < 8; ++r) {
        const int row = rt * 16 + hf * 8 + r;
        out[((size_t)m * N_ + n0 + row) * D_ + col] = acc[rt][ct][r];
      }
    }
}

// ---------------------------------------------------------------------------
extern "C" void kernel_launch(void* const* d_in, const int* in_sizes, int n_in,
                              void* d_out, int out_size, void* d_ws, size_t ws_size,
                              hipStream_t stream) {
  (void)in_sizes; (void)n_in; (void)out_size; (void)ws_size;

  const float* video = (const float*)d_in[0];
  const float* music = (const float*)d_in[1];
  const int*   mask  = (const int*)d_in[2];
  const float* ln1_g = (const float*)d_in[3];
  const float* ln1_b = (const float*)d_in[4];
  const float* Wq = (const float*)d_in[5];  const float* bq = (const float*)d_in[6];
  const float* Wk = (const float*)d_in[7];  const float* bk = (const float*)d_in[8];
  const float* Wv = (const float*)d_in[9];  const float* bv = (const float*)d_in[10];
  const float* Wo = (const float*)d_in[11]; const float* bo = (const float*)d_in[12];
  const float* Wl = (const float*)d_in[13]; const float* bl = (const float*)d_in[14];
  const float* ln2_g = (const float*)d_in[15]; const float* ln2_b = (const float*)d_in[16];
  const float* ln3_g = (const float*)d_in[17]; const float* ln3_b = (const float*)d_in[18];
  float* out = (float*)d_out;

  // workspace carve-out (~195 MB)
  char* ws = (char*)d_ws;
  size_t off = 0;
  auto alloc = [&](size_t bytes) -> void* {
    void* p = ws + off;
    off = (off + bytes + 255) & ~(size_t)255;
    return p;
  };
  __bf16* mln = (__bf16*)alloc((size_t)M_ * S_ * D_ * 2);
  __bf16* vln = (__bf16*)alloc((size_t)N_ * D_ * 2);
  __bf16* qb  = (__bf16*)alloc((size_t)N_ * D_ * 2);
  __bf16* kb  = (__bf16*)alloc((size_t)M_ * S_ * D_ * 2);
  __bf16* vtb = (__bf16*)alloc((size_t)M_ * D_ * S_ * 2);  // transposed V
  __bf16* Wq_b = (__bf16*)alloc((size_t)D_ * D_ * 2);
  __bf16* Wk_b = (__bf16*)alloc((size_t)D_ * D_ * 2);
  __bf16* Wv_b = (__bf16*)alloc((size_t)D_ * D_ * 2);
  __bf16* Wo_b = (__bf16*)alloc((size_t)D_ * D_ * 2);
  __bf16* Wl_b = (__bf16*)alloc((size_t)D_ * D_ * 2);

  // LN1 (both streams) -> bf16
  ln_to_bf16<<<N_, 256, 0, stream>>>(video, ln1_g, ln1_b, vln);
  ln_to_bf16<<<M_ * S_, 256, 0, stream>>>(music, ln1_g, ln1_b, mln);

  // weight conversion
  const int wn = D_ * D_;
  cvt_bf16<<<wn / 256, 256, 0, stream>>>(Wq, Wq_b, wn);
  cvt_bf16<<<wn / 256, 256, 0, stream>>>(Wk, Wk_b, wn);
  cvt_bf16<<<wn / 256, 256, 0, stream>>>(Wv, Wv_b, wn);
  cvt_bf16<<<wn / 256, 256, 0, stream>>>(Wo, Wo_b, wn);
  cvt_bf16<<<wn / 256, 256, 0, stream>>>(Wl, Wl_b, wn);

  // projections (WMMA GEMMs); V written transposed per music item
  gemm_xwT<<<N_ / 32, 256, 0, stream>>>(vln, Wq_b, bq, qb, 0);
  gemm_xwT<<<(M_ * S_) / 32, 256, 0, stream>>>(mln, Wk_b, bk, kb, 0);
  gemm_xwT<<<(M_ * S_) / 32, 256, 0, stream>>>(mln, Wv_b, bv, vtb, 1);

  // fused attention + output head
  dim3 grid(M_, N_ / 32);
  attn_fused<<<grid, 256, 0, stream>>>(qb, kb, vtb, mask, Wo_b, bo, Wl_b, bl,
                                       ln2_g, ln2_b, ln3_g, ln3_b, out);
}
